// VQExpert_52347061403783
// MI455X (gfx1250) — matrix operand order repaired and checked by reading, not compiled, not run on hardware
//
#include <hip/hip_runtime.h>

typedef _Float16 f16;
typedef __attribute__((ext_vector_type(16))) _Float16 v16h;
typedef __attribute__((ext_vector_type(8)))  _Float16 v8h;
typedef __attribute__((ext_vector_type(8)))  float    v8f;

// LDS layout: all weights converted to f16 once per block; per-wave bounce
// buffers for the D-layout -> A-layout transpose between chained GEMMs.
struct __align__(32) SM {
    f16 wdown[128 * 128];   // [out][in] row-major, f16
    f16 wup[128 * 128];     // [out][in]
    f16 wpin[32 * 128];     // [out=32][in=128]
    f16 wpout[128 * 32];    // [out=128][in=32]
    f16 cbh[256 * 32];      // codebook as f16, [code][dim]
    f16 bounce[8][16 * 128];// per-wave activation bounce tile (16 rows x 128)
    float cnorm[256];       // -0.5 * ||codebook[c]||^2  (fp32, from fp32 source)
    float bdown[128], bpout[128], bup[128], bpin[32];
    int   idxw[8][16];      // per-wave argmin indices for its 16 rows
};

#define WMMA_F16(a, b, c) \
    __builtin_amdgcn_wmma_f32_16x16x32_f16(false, (a), false, (b), (short)0, (c), false, false)

__global__ void __launch_bounds__(256) vq_expert_kernel(
    const float* __restrict__ x,
    const float* __restrict__ Wd,  const float* __restrict__ bd,
    const float* __restrict__ Wpi, const float* __restrict__ bpi,
    const float* __restrict__ cb,
    const float* __restrict__ Wpo, const float* __restrict__ bpo,
    const float* __restrict__ Wu,  const float* __restrict__ bu,
    float* __restrict__ y, int* __restrict__ indices, float* __restrict__ commit,
    int ntiles)
{
    extern __shared__ __align__(32) char smem_raw[];
    SM* s = reinterpret_cast<SM*>(smem_raw);
    const int t = threadIdx.x;

    // ---- stage weights to LDS (f16) ----
    for (int i = t; i < 128 * 128; i += 256) { s->wdown[i] = (f16)Wd[i]; s->wup[i] = (f16)Wu[i]; }
    for (int i = t; i < 32 * 128;  i += 256) { s->wpin[i]  = (f16)Wpi[i]; s->wpout[i] = (f16)Wpo[i]; }
    for (int i = t; i < 256 * 32;  i += 256) { s->cbh[i]   = (f16)cb[i]; }
    if (t < 128) { s->bdown[t] = bd[t]; s->bpout[t] = bpo[t]; s->bup[t] = bu[t]; }
    if (t < 32)  { s->bpin[t] = bpi[t]; }
    {   // one codebook row per thread (256 threads == 256 codes)
        const float* c = cb + t * 32;
        float acc = 0.f;
        #pragma unroll
        for (int k = 0; k < 32; k++) acc += c[k] * c[k];
        s->cnorm[t] = -0.5f * acc;
    }
    __syncthreads();
    if (blockIdx.x == 0 && t == 0) commit[0] = 0.f;   // commitment_weight == 0

    const int  lane  = t & 31;
    const int  wid   = t >> 5;
    const int  mrow  = lane & 15;
    const bool hiH   = lane >= 16;
    const int  rbase = hiH ? 8 : 0;       // C/D rows: vgpr r -> row r (+8 upper half)
    const int  kh    = hiH ? 8 : 0;       // A layout half-K offset
    f16* hb = s->bounce[wid];

    // B fragment (32x16, f16): lane -> column, K halves {kh..kh+7, 16+kh..16+kh+7} per lane.
    auto loadB = [&](const f16* W, int ks, int nb, int kb) -> v16h {
        const f16* p = W + (nb * 16 + mrow) * ks + kb * 32 + kh;
        v8h lo = *(const v8h*)p, hc = *(const v8h*)(p + 16);
        v16h b;
        #pragma unroll
        for (int j = 0; j < 8; j++) { b[j] = lo[j]; b[8 + j] = hc[j]; }
        return b;
    };
    // A fragment (16x32, f16) from a row-major f16 LDS tile.
    auto loadA = [&](const f16* T, int ks, int kb) -> v16h {
        const f16* p = T + mrow * ks + kb * 32 + kh;
        v8h lo = *(const v8h*)p, hc = *(const v8h*)(p + 16);
        v16h a;
        #pragma unroll
        for (int j = 0; j < 8; j++) { a[j] = lo[j]; a[8 + j] = hc[j]; }
        return a;
    };
    // C init: bias is per output column -> same value in all 8 row-vgprs.
    auto biasC = [&](const float* b, int nb) -> v8f {
        float bvv = b[nb * 16 + mrow];
        v8f c;
        #pragma unroll
        for (int r = 0; r < 8; r++) c[r] = bvv;
        return c;
    };
    // D fragment -> row-major f16 LDS tile.
    auto storeD = [&](f16* T, int ks, int nb, v8f d) {
        #pragma unroll
        for (int r = 0; r < 8; r++) T[(rbase + r) * ks + nb * 16 + mrow] = (f16)d[r];
    };

    // Only the K-loop-free GEMM3 weights stay resident in VGPRs; everything
    // else is re-read from LDS to keep the wave under 256 VGPRs (avoids
    // s_set_vgpr_msb churn seen with larger hoisted sets).
    v16h Bpout[8];
    #pragma unroll
    for (int nb = 0; nb < 8; nb++) Bpout[nb] = loadB(s->wpout, 32, nb, 0);

    const int stride = gridDim.x * 8;
    for (int tile = blockIdx.x * 8 + wid; tile < ntiles; tile += stride) {
        const int row0 = tile * 16;
        const float* xr = x + (size_t)(row0 + mrow) * 128;
        {   // prefetch next tile's rows (global_prefetch_b8)
            int nt = tile + stride;
            if (nt < ntiles) __builtin_prefetch(x + (size_t)(nt * 16 + mrow) * 128, 0, 3);
        }

        // A fragments of x (fp32 global -> f16 A layout)
        v16h ax[4];
        #pragma unroll
        for (int kb = 0; kb < 4; kb++) {
            const float* p = xr + kb * 32 + kh;
            v16h a;
            #pragma unroll
            for (int j = 0; j < 8; j++) { a[j] = (f16)p[j]; a[8 + j] = (f16)p[16 + j]; }
            ax[kb] = a;
        }

        // ---- GEMM1: h = x @ Wdown^T + bdown  (16x128) ----
        #pragma unroll
        for (int nb = 0; nb < 8; nb++) {
            v8f acc = biasC(s->bdown, nb);
            #pragma unroll
            for (int kb = 0; kb < 4; kb++) acc = WMMA_F16(ax[kb], loadB(s->wdown, 128, nb, kb), acc);
            storeD(hb, 128, nb, acc);
        }
        v16h ah[4];
        #pragma unroll
        for (int kb = 0; kb < 4; kb++) ah[kb] = loadA(hb, 128, kb);

        // ---- GEMM2: z = h @ Wpin^T + bpin  (16x32) ----
        #pragma unroll
        for (int nb = 0; nb < 2; nb++) {
            v8f acc = biasC(s->bpin, nb);
            #pragma unroll
            for (int kb = 0; kb < 4; kb++) acc = WMMA_F16(ah[kb], loadB(s->wpin, 128, nb, kb), acc);
            storeD(hb, 32, nb, acc);
        }
        v16h az = loadA(hb, 32, 0);

        // ---- scores = z . c - 0.5||c||^2 ; argmax == argmin d2 ----
        float bv[8]; int bi[8];
        #pragma unroll
        for (int r = 0; r < 8; r++) { bv[r] = -3.4e38f; bi[r] = 0; }
        #pragma unroll
        for (int ct = 0; ct < 16; ct++) {
            float cn = s->cnorm[ct * 16 + mrow];
            v8f c;
            #pragma unroll
            for (int r = 0; r < 8; r++) c[r] = cn;
            v8f sc = WMMA_F16(az, loadB(s->cbh, 32, ct, 0), c);
            const int ci = ct * 16 + mrow;
            #pragma unroll
            for (int r = 0; r < 8; r++)
                if (sc[r] > bv[r]) { bv[r] = sc[r]; bi[r] = ci; }   // ascending ci: first-min tiebreak
        }
        // butterfly over the 16 lanes holding each row's columns
        #pragma unroll
        for (int m = 1; m < 16; m <<= 1) {
            #pragma unroll
            for (int r = 0; r < 8; r++) {
                float ov = __shfl_xor(bv[r], m, 32);
                int   oi = __shfl_xor(bi[r], m, 32);
                if (ov > bv[r] || (ov == bv[r] && oi < bi[r])) { bv[r] = ov; bi[r] = oi; }
            }
        }
        if (mrow == 0) {
            #pragma unroll
            for (int r = 0; r < 8; r++) {
                s->idxw[wid][rbase + r] = bi[r];
                indices[row0 + rbase + r] = bi[r];
            }
        }
        // gather q = codebook[idx] straight into an A fragment (STE forward == q)
        int myidx = s->idxw[wid][mrow];
        const f16* qp = s->cbh + myidx * 32 + kh;
        v16h aq;
        {
            v8h lo = *(const v8h*)qp, hc = *(const v8h*)(qp + 16);
            #pragma unroll
            for (int j = 0; j < 8; j++) { aq[j] = lo[j]; aq[8 + j] = hc[j]; }
        }

        // ---- GEMM3: v = q @ Wpout^T + bpout  (16x128) ----
        #pragma unroll
        for (int nb = 0; nb < 8; nb++) {
            v8f acc = biasC(s->bpout, nb);
            acc = WMMA_F16(aq, Bpout[nb], acc);
            storeD(hb, 128, nb, acc);
        }
        v16h av[4];
        #pragma unroll
        for (int kb = 0; kb < 4; kb++) av[kb] = loadA(hb, 128, kb);

        // ---- GEMM4: y = clamp(v @ Wup^T + bup, -1, 1) ----
        #pragma unroll
        for (int nb = 0; nb < 8; nb++) {
            v8f acc = biasC(s->bup, nb);
            #pragma unroll
            for (int kb = 0; kb < 4; kb++) acc = WMMA_F16(av[kb], loadB(s->wup, 128, nb, kb), acc);
            #pragma unroll
            for (int r = 0; r < 8; r++) {
                float v = fminf(fmaxf(acc[r], -1.f), 1.f);
                y[(size_t)(row0 + rbase + r) * 128 + nb * 16 + mrow] = v;
            }
        }
    }
}

extern "C" void kernel_launch(void* const* d_in, const int* in_sizes, int n_in,
                              void* d_out, int out_size, void* d_ws, size_t ws_size,
                              hipStream_t stream) {
    const float* x   = (const float*)d_in[0];
    const float* Wd  = (const float*)d_in[1];
    const float* bd  = (const float*)d_in[2];
    const float* Wpi = (const float*)d_in[3];
    const float* bpi = (const float*)d_in[4];
    const float* cb  = (const float*)d_in[5];
    const float* Wpo = (const float*)d_in[6];
    const float* bpo = (const float*)d_in[7];
    const float* Wu  = (const float*)d_in[8];
    const float* bu  = (const float*)d_in[9];

    const int N = in_sizes[0] / 128;          // 500000
    float* yout   = (float*)d_out;                            // [N,128] f32
    int*   idxout = (int*)(yout + (size_t)N * 128);           // [N] i32
    float* commit = (float*)(yout + (size_t)N * 128 + N);     // scalar

    const int ntiles = N / 16;                // N % 16 == 0 for this workload
    dim3 grid(512), block(256);
    vq_expert_kernel<<<grid, block, sizeof(SM), stream>>>(
        x, Wd, bd, Wpi, bpi, cb, Wpo, bpo, Wu, bu, yout, idxout, commit, ntiles);
}
